// GNNEncoder_38766374814022
// MI455X (gfx1250) — matrix operand order repaired and checked by reading, not compiled, not run on hardware
//
#include <hip/hip_runtime.h>
#include <hip/hip_bf16.h>

typedef __attribute__((ext_vector_type(16))) _Float16 v16h;
typedef __attribute__((ext_vector_type(8)))  float    v8f;

#define N_NODES 50000
#define N_EDGES 1600000
#define HID 64

// ---------------------------------------------------------------------------
// Utility kernels
// ---------------------------------------------------------------------------
__global__ void zero_f32(float* __restrict__ p, int n) {
    int i = blockIdx.x * blockDim.x + threadIdx.x;
    if (i < n) p[i] = 0.0f;
}

__global__ void count_edges(const int* __restrict__ dst, float* __restrict__ cnt) {
    int e = blockIdx.x * blockDim.x + threadIdx.x;
    if (e < N_EDGES) atomicAdd(&cnt[dst[e]], 1.0f);
}

__global__ void make_inv(float* __restrict__ cnt) {
    int i = blockIdx.x * blockDim.x + threadIdx.x;
    if (i < N_NODES) cnt[i] = 1.0f / fmaxf(cnt[i], 1.0f);
}

// Scatter-add of 64-channel messages: 16 lanes per edge, 4 channels per lane.
// P[src] read as coalesced float4; accumulator A (12.8 MB) is L2-resident
// (192 MB L2) so the fp32 atomics resolve at the L2 atomic units.
__global__ void scatter_add64(const float* __restrict__ P,
                              const int* __restrict__ src,
                              const int* __restrict__ dst,
                              float* __restrict__ A) {
    int t  = blockIdx.x * blockDim.x + threadIdx.x;
    int e  = t >> 4;
    int c4 = (t & 15) << 2;
    if (e < N_EDGES) {
        int s = src[e];
        int d = dst[e];
        const float4 v = *(const float4*)(P + s * HID + c4);
        float* a = A + d * HID + c4;
        atomicAdd(a + 0, v.x);
        atomicAdd(a + 1, v.y);
        atomicAdd(a + 2, v.z);
        atomicAdd(a + 3, v.w);
    }
}

// hout = maybe_relu(A * inv_cnt + bias + S)
__global__ void combine64(const float* __restrict__ A,
                          const float* __restrict__ S,
                          const float* __restrict__ inv,
                          const float* __restrict__ bias,
                          float* __restrict__ out, int relu) {
    int t = blockIdx.x * blockDim.x + threadIdx.x;
    if (t < N_NODES * HID) {
        int node = t >> 6;
        int c    = t & 63;
        float v = A[t] * inv[node] + bias[c] + S[t];
        out[t] = relu ? fmaxf(v, 0.0f) : v;
    }
}

// ---------------------------------------------------------------------------
// WMMA GEMM: C[N, DOUT] = Ain[N, DIN] @ W[DIN, DOUT] (+ bias), fp32 in/out,
// f16 operands, f32 accumulate (v_wmma_f32_16x16x32_f16).
//
// blockDim = (DOUT/16)*32: each wave owns one 16-col tile and preloads ALL
// B fragments for its column tile into registers (DIN/32 fragments, <=32
// VGPRs), then sweeps RT row tiles (grid.x * RT * 16 == N rows, exact),
// loading A fragments as 4x b128 and issuing DIN/32 WMMAs per tile.
// All strides compile-time -> int saddr+voffset addressing, no u64 muls.
// EXEC is all-1s at every WMMA (no divergence, blockDim multiple of 32).
// ---------------------------------------------------------------------------
template <int DIN, int DOUT, int RT, bool HAS_BIAS>
__global__ void wmma_gemm(const float* __restrict__ Ain,
                          const float* __restrict__ W,
                          const float* __restrict__ bias,
                          float* __restrict__ C) {
    constexpr int KSTEPS = DIN / 32;
    const int lane = threadIdx.x & 31;
    const int wave = threadIdx.x >> 5;   // column tile index
    const int m    = lane & 15;
    const int hi   = lane >> 4;
    const int col  = (wave << 4) + m;    // B/C/D column

    // ---- B fragments (32x16 f16): per lane, column `col`, K = kk+16*hi+2i(+1)
    v16h bfrag[KSTEPS];
#pragma unroll
    for (int ks = 0; ks < KSTEPS; ++ks) {
        const int kb = ks * 32 + 16 * hi;
#pragma unroll
        for (int i = 0; i < 8; ++i) {
            bfrag[ks][2 * i]     = (_Float16)W[(kb + 2 * i)     * DOUT + col];
            bfrag[ks][2 * i + 1] = (_Float16)W[(kb + 2 * i + 1) * DOUT + col];
        }
    }
    const float bv = HAS_BIAS ? bias[col] : 0.0f;

    int row0 = blockIdx.x * (16 * RT);
#pragma unroll
    for (int rt = 0; rt < RT; ++rt, row0 += 16) {
        v8f c = {};
#pragma unroll
        for (int ks = 0; ks < KSTEPS; ++ks) {
            // A fragment (16x32 f16): elements 0..15 = row, K = kk+16*hi+0..15
            // -> one contiguous 16-float run: 4x float4 (global_load_b128).
            const float* ap = Ain + (row0 + m) * DIN + ks * 32 + 16 * hi;
            const float4 q0 = ((const float4*)ap)[0];
            const float4 q1 = ((const float4*)ap)[1];
            const float4 q2 = ((const float4*)(ap + 8))[0];
            const float4 q3 = ((const float4*)(ap + 8))[1];
            v16h a;
            a[0]  = (_Float16)q0.x;  a[1]  = (_Float16)q0.y;
            a[2]  = (_Float16)q0.z;  a[3]  = (_Float16)q0.w;
            a[4]  = (_Float16)q1.x;  a[5]  = (_Float16)q1.y;
            a[6]  = (_Float16)q1.z;  a[7]  = (_Float16)q1.w;
            a[8]  = (_Float16)q2.x;  a[9]  = (_Float16)q2.y;
            a[10] = (_Float16)q2.z;  a[11] = (_Float16)q2.w;
            a[12] = (_Float16)q3.x;  a[13] = (_Float16)q3.y;
            a[14] = (_Float16)q3.z;  a[15] = (_Float16)q3.w;

            c = __builtin_amdgcn_wmma_f32_16x16x32_f16(
                    /*neg_a=*/false, a, /*neg_b=*/false, bfrag[ks],
                    /*c_mod=*/(short)0, c, /*reuse_a=*/false, /*reuse_b=*/false);
        }
        // D fragment: VGPR r holds row M = r + 8*hi, column = col.
#pragma unroll
        for (int r = 0; r < 8; ++r) {
            C[(row0 + r + 8 * hi) * DOUT + col] = c[r] + bv;
        }
    }
}

// ---------------------------------------------------------------------------
// Host-side orchestration
// ---------------------------------------------------------------------------
extern "C" void kernel_launch(void* const* d_in, const int* in_sizes, int n_in,
                              void* d_out, int out_size, void* d_ws, size_t ws_size,
                              hipStream_t stream) {
    const float* x    = (const float*)d_in[0];
    const int*   ei   = (const int*)d_in[1];
    const int*   src  = ei;
    const int*   dst  = ei + N_EDGES;

    const float* W1a_l = (const float*)d_in[2];
    const float* b1a   = (const float*)d_in[3];
    const float* W1a_r = (const float*)d_in[4];
    const float* W1b_l = (const float*)d_in[5];
    const float* b1b   = (const float*)d_in[6];
    const float* W1b_r = (const float*)d_in[7];
    const float* W2a_l = (const float*)d_in[8];
    const float* b2a   = (const float*)d_in[9];
    const float* W2a_r = (const float*)d_in[10];
    const float* W2b_l = (const float*)d_in[11];
    const float* b2b   = (const float*)d_in[12];
    const float* W2b_r = (const float*)d_in[13];
    const float* Wlin  = (const float*)d_in[14];
    const float* blin  = (const float*)d_in[15];

    // Workspace layout (floats): inv[N] | P[N*64] | S[N*64] | A[N*64] | h1 | h2
    float* inv = (float*)d_ws;
    float* P   = inv + N_NODES;
    float* S   = P + (size_t)N_NODES * HID;
    float* A   = S + (size_t)N_NODES * HID;
    float* h1  = A + (size_t)N_NODES * HID;
    float* h2  = h1 + (size_t)N_NODES * HID;

    const int NHID = N_NODES * HID;
    const dim3 blk256(256);
    const dim3 g_nodes((N_NODES + 255) / 256);
    const dim3 g_edges((N_EDGES + 255) / 256);
    const dim3 g_scat((N_EDGES * 16 + 255) / 256);
    const dim3 g_nhid((NHID + 255) / 256);
    // 50000 rows = 625 blocks * 5 row-tiles * 16
    const dim3 g_gemm(N_NODES / (16 * 5));

    // Degree -> reciprocal (edges are constant across layers; compute once)
    zero_f32<<<g_nodes, blk256, 0, stream>>>(inv, N_NODES);
    count_edges<<<g_edges, blk256, 0, stream>>>(dst, inv);
    make_inv<<<g_nodes, blk256, 0, stream>>>(inv);

    auto sage = [&](const float* hin, bool first, const float* Wl, const float* bl,
                    const float* Wr, float* hout, int relu) {
        // Transform first (linearity of mean): halves edge traffic for layer 1
        if (first) {
            wmma_gemm<128, HID, 5, false><<<g_gemm, dim3(128), 0, stream>>>(hin, Wl, nullptr, P);
            wmma_gemm<128, HID, 5, false><<<g_gemm, dim3(128), 0, stream>>>(hin, Wr, nullptr, S);
        } else {
            wmma_gemm<HID, HID, 5, false><<<g_gemm, dim3(128), 0, stream>>>(hin, Wl, nullptr, P);
            wmma_gemm<HID, HID, 5, false><<<g_gemm, dim3(128), 0, stream>>>(hin, Wr, nullptr, S);
        }
        zero_f32<<<g_nhid, blk256, 0, stream>>>(A, NHID);
        scatter_add64<<<g_scat, blk256, 0, stream>>>(P, src, dst, A);
        combine64<<<g_nhid, blk256, 0, stream>>>(A, S, inv, bl, hout, relu);
    };

    sage(x,  true,  W1a_l, b1a, W1a_r, h1, 1);
    sage(h1, false, W1b_l, b1b, W1b_r, h2, 1);
    sage(h2, false, W2a_l, b2a, W2a_r, h1, 1);
    sage(h1, false, W2b_l, b2b, W2b_r, h2, 0);

    // Final linear: [N,64] @ [64,32] + blin -> d_out  (2 waves per block)
    wmma_gemm<HID, 32, 5, true><<<g_gemm, dim3(64), 0, stream>>>(h2, Wlin, blin, (float*)d_out);
}
